// ModulatedConv2d_22325240005067
// MI455X (gfx1250) — matrix operand order repaired and checked by reading, not compiled, not run on hardware
//
#include <hip/hip_runtime.h>
#include <hip/hip_bf16.h>

typedef __bf16 bf16;
typedef bf16  v16bf __attribute__((ext_vector_type(16)));
typedef bf16  v8bf  __attribute__((ext_vector_type(8)));
typedef float v8f   __attribute__((ext_vector_type(8)));
typedef unsigned int u32x4 __attribute__((ext_vector_type(4)));
typedef int i32x8 __attribute__((ext_vector_type(8)));
typedef int i32x4 __attribute__((ext_vector_type(4)));

#define W_DIM   512
#define IN_CH   512
#define OUT_CH  512
#define HW      64
#define NPIX    4096            // 64*64
#define KTAPS   9
#define KSTRIDE 40              // padded k-stride (80B): conflict-free b128 reads
#define XPIX    416             // 6 rows x 66 cols = 396, padded to 13*32
#define NKC     16              // 512/32 K-chunks
#define NTILES  (NKC * KTAPS)   // 144 A tiles
#define WEIGHT_GAIN     (1.0f / 67.88225099390856f)   // 1/sqrt(512*9)
#define MOD_WEIGHT_GAIN (1.0f / 22.627416997969522f)  // 1/sqrt(512)

union Frag { v16bf v; v8bf h[2]; };

// ---------------- small prep kernels ----------------

__global__ void mc_style_kernel(const float* __restrict__ w,
                                const float* __restrict__ mod_w,
                                const float* __restrict__ mod_b,
                                float* __restrict__ style) {
    int idx = blockIdx.x * blockDim.x + threadIdx.x;   // 16*512
    int b = idx >> 9, i = idx & 511;
    const float* wr = w + b * W_DIM;
    const float* mw = mod_w + i * W_DIM;
    float acc = 0.f;
    #pragma unroll 4
    for (int j = 0; j < W_DIM; ++j) acc += wr[j] * mw[j];
    style[idx] = acc * MOD_WEIGHT_GAIN + mod_b[i];
}

__global__ void mc_s2_kernel(const float* __restrict__ weight,
                             float* __restrict__ s2) {
    int idx = blockIdx.x * blockDim.x + threadIdx.x;   // 512*512: o*512+i
    const float* wp = weight + (size_t)idx * KTAPS;
    float s = 0.f;
    #pragma unroll
    for (int k = 0; k < KTAPS; ++k) {
        float v = wp[k] * WEIGHT_GAIN;
        s += v * v;
    }
    s2[idx] = s;
}

__global__ void mc_demod_kernel(const float* __restrict__ style,
                                const float* __restrict__ s2,
                                float* __restrict__ demod) {
    int idx = blockIdx.x * blockDim.x + threadIdx.x;   // 16*512: b*512+o
    int b = idx >> 9, o = idx & 511;
    const float* st = style + b * IN_CH;
    const float* sp = s2 + o * IN_CH;
    float s = 0.f;
    #pragma unroll 4
    for (int i = 0; i < IN_CH; ++i) {
        float v = st[i];
        s += v * v * sp[i];
    }
    demod[idx] = rsqrtf(s + 1e-8f);
}

// Repack weight[o][i][tap] (f32) -> Wr[tap][o][i] (bf16, pre-scaled by gain)
__global__ void mc_repack_kernel(const float* __restrict__ weight,
                                 bf16* __restrict__ Wr) {
    int idx = blockIdx.x * blockDim.x + threadIdx.x;   // 9*512*512
    int tap = idx / (OUT_CH * IN_CH);
    int rem = idx - tap * (OUT_CH * IN_CH);
    int o = rem >> 9, i = rem & 511;
    Wr[idx] = (bf16)(weight[((size_t)o * IN_CH + i) * KTAPS + tap] * WEIGHT_GAIN);
}

// ---------------- TDM: DMA one A tile (128 rows x 32 bf16) into LDS ----------
// Tensor: 512x512 bf16 (row stride 512). Tile: dim0=32 (i), dim1=128 (o).
// LDS pad: +4 DWORDs (16B) after every 16 DWORDs (64B) -> 80B row pitch = KSTRIDE.
__device__ __forceinline__ void tdm_load_A(const bf16* gsrc, unsigned lds_addr) {
    unsigned long long ga = (unsigned long long)(uintptr_t)gsrc;
    u32x4 g0;
    g0[0] = 1u;                                            // count=1 (valid D#)
    g0[1] = lds_addr;                                      // LDS byte address
    g0[2] = (unsigned)(ga & 0xFFFFFFFFu);                  // global_addr[31:0]
    g0[3] = (unsigned)((ga >> 32) & 0x1FFFFFFu)            // global_addr[56:32]
          | (2u << 30);                                    // type=2 ("image")
    i32x8 g1;
    g1[0] = (int)((1u << 16)     // data_size=1 -> 2 bytes
                | (1u << 20)     // pad_enable
                | (3u << 22)     // pad_interval: 16 DWORDs
                | (3u << 25));   // pad_amount:   4 DWORDs
    g1[1] = (int)(512u << 16);                 // tensor_dim0[15:0] in hi half
    g1[2] = (int)(512u << 16);                 // td0 hi=0 | tensor_dim1 lo
    g1[3] = (int)(32u << 16);                  // td1 hi=0 | tile_dim0=32
    g1[4] = (int)128u;                         // tile_dim1=128 | tile_dim2=0
    g1[5] = (int)512;                          // tensor_dim0_stride lo32
    g1[6] = 0;                                 // stride hi | dim1_stride lo
    g1[7] = 0;
    i32x4 z4 = {0, 0, 0, 0};                   // 2-D tensor: groups 2/3 unused
    i32x8 z8 = {0, 0, 0, 0, 0, 0, 0, 0};
    __builtin_amdgcn_tensor_load_to_lds(g0, g1, z4, z4, z8, 0);
}

// ---------------- main implicit-GEMM conv ----------------
// grid: (16 pixel tiles of 256, 4 o-tiles of 128, 16 batch), block 256 (8 waves)
// WG tile 128(M) x 256(N); wave tile 64x64 -> 16 WMMAs per tap.
// kc outer (stage haloed x once), tap inner (reuse x tile 9x).
// A tiles fetched by the Tensor Data Mover, double-buffered, DMA depth 1.
__global__ __launch_bounds__(256)
void mc_conv_kernel(const float* __restrict__ x,
                    const bf16*  __restrict__ Wr,
                    const float* __restrict__ style,
                    const float* __restrict__ demod,
                    float* __restrict__ out) {
    __shared__ bf16 ldsX[XPIX * KSTRIDE];        // ~32.5 KB, [pixel q][k]
    __shared__ bf16 ldsA[2][128 * KSTRIDE];      // 2 x ~10 KB, [o-row][k]

    const int b     = blockIdx.z;
    const int om    = blockIdx.y * 128;
    const int pbase = blockIdx.x * 256;          // four image rows
    const int h0    = blockIdx.x * 4;

    const int tid  = threadIdx.x;
    const int wave = tid >> 5;
    const int lane = tid & 31;
    const int wm   = wave >> 2;                  // 0..1 (64 rows each)
    const int wn   = wave & 3;                   // 0..3 (64 cols each)
    const int lrow = lane & 15;
    const int ahalf = (lane & 16) ? 8 : 0;       // A frag K split
    const int bhalf = (lane & 16) ? 8 : 0;       // B frag K split

    const float* styleRow = style + b * IN_CH;
    const float* xb       = x + (size_t)b * IN_CH * NPIX;

    v8f acc[4][4];
    #pragma unroll
    for (int mt = 0; mt < 4; ++mt)
        #pragma unroll
        for (int nt = 0; nt < 4; ++nt)
            acc[mt][nt] = (v8f){0.f, 0.f, 0.f, 0.f, 0.f, 0.f, 0.f, 0.f};

    const int wch = wave * 4;                    // X-stage: 4 channels per wave

    // prologue: kick off first A tile (tap 0, kc 0)
    if (wave == 0)
        tdm_load_A(Wr + (size_t)om * IN_CH,
                   (unsigned)(uintptr_t)&ldsA[0][0]);

    for (int kc = 0; kc < IN_CH; kc += 32) {
        __syncthreads();   // ldsX safe to rewrite (prev kc readers done)

        // ---- stage haloed X patch: 396 pixels x 32 ch, style folded in ----
        #pragma unroll
        for (int p = 0; p < 2; ++p) {
            const int kl  = wch + 2 * p;         // local k (even)
            const int ch  = kc + kl;
            const float s0 = styleRow[ch];
            const float s1 = styleRow[ch + 1];
            const float* xc0 = xb + (size_t)ch * NPIX;
            const float* xc1 = xc0 + NPIX;
            #pragma unroll
            for (int i = 0; i < 13; ++i) {
                int q    = i * 32 + lane;        // 0..415 (>=396 unused)
                int prow = q / 66;
                int pcol = q - prow * 66;
                int sy   = h0 - 1 + prow;
                int sx   = pcol - 1;
                int syc = min(max(sy, 0), HW - 1);
                int sxc = min(max(sx, 0), HW - 1);
                int off = syc * HW + sxc;
                float v0 = xc0[off];
                float v1 = xc1[off];
                bool inb = ((unsigned)sy < (unsigned)HW) &
                           ((unsigned)sx < (unsigned)HW);
                v0 = inb ? v0 * s0 : 0.f;
                v1 = inb ? v1 * s1 : 0.f;
                union { bf16 h[2]; unsigned u; } pk;
                pk.h[0] = (bf16)v0;
                pk.h[1] = (bf16)v1;
                *(unsigned*)&ldsX[q * KSTRIDE + kl] = pk.u;
            }
        }

        for (int tap = 0; tap < KTAPS; ++tap) {
            const int t  = (kc >> 5) * KTAPS + tap;    // 0..143
            const int dy = tap / 3 - 1;
            const int dx = tap % 3 - 1;

            if (wave == 0)
                __builtin_amdgcn_s_wait_tensorcnt(0);  // buf[t&1] landed
            __syncthreads();

            // issue next tile into the other buffer (overlaps with WMMAs)
            if (wave == 0) {
                int tn = t + 1;
                if (tn < NTILES) {
                    int tapn = tn % KTAPS;
                    int kcn  = (tn / KTAPS) << 5;
                    tdm_load_A(Wr + (size_t)tapn * OUT_CH * IN_CH
                                  + (size_t)om * IN_CH + kcn,
                               (unsigned)(uintptr_t)&ldsA[tn & 1][0]);
                }
            }

            // ---- fragments + 16 WMMAs from buf[t&1] ----
            const bf16* At = &ldsA[t & 1][0];
            Frag afr[4], bfr[4];
            #pragma unroll
            for (int mt = 0; mt < 4; ++mt) {
                int r = (wm * 64 + mt * 16 + lrow) * KSTRIDE;
                afr[mt].h[0] = *(const v8bf*)&At[r + ahalf];
                afr[mt].h[1] = *(const v8bf*)&At[r + ahalf + 16];
            }
            #pragma unroll
            for (int nt = 0; nt < 4; ++nt) {
                int n = wn * 64 + nt * 16 + lrow;      // 0..255
                int r = n >> 6;
                int c = n & 63;
                int q = (r + dy + 1) * 66 + (c + dx + 1);
                bfr[nt].h[0] = *(const v8bf*)&ldsX[q * KSTRIDE + bhalf];
                bfr[nt].h[1] = *(const v8bf*)&ldsX[q * KSTRIDE + bhalf + 16];
            }
            #pragma unroll
            for (int mt = 0; mt < 4; ++mt)
                #pragma unroll
                for (int nt = 0; nt < 4; ++nt)
                    acc[mt][nt] = __builtin_amdgcn_wmma_f32_16x16x32_bf16(
                        false, afr[mt].v, false, bfr[nt].v,
                        (short)0, acc[mt][nt], false, false);
            // no trailing barrier: next iteration's wait+barrier covers WAR
        }
    }

    // ---- epilogue: demod scale + store f32 ----
    const float* demodRow = demod + b * OUT_CH;
    const int mofs = (lane & 16) ? 8 : 0;
    #pragma unroll
    for (int mt = 0; mt < 4; ++mt) {
        #pragma unroll
        for (int nt = 0; nt < 4; ++nt) {
            int n = wn * 64 + nt * 16 + lrow;
            int p = pbase + n;
            #pragma unroll
            for (int r = 0; r < 8; ++r) {
                int o = om + wm * 64 + mt * 16 + r + mofs;
                out[((size_t)b * OUT_CH + o) * NPIX + p] = acc[mt][nt][r] * demodRow[o];
            }
        }
    }
}

// ---------------- launcher ----------------

extern "C" void kernel_launch(void* const* d_in, const int* in_sizes, int n_in,
                              void* d_out, int out_size, void* d_ws, size_t ws_size,
                              hipStream_t stream) {
    (void)in_sizes; (void)n_in; (void)out_size; (void)ws_size;
    const float* x        = (const float*)d_in[0];   // [16,512,64,64]
    const float* w        = (const float*)d_in[1];   // [16,512]
    const float* weight   = (const float*)d_in[2];   // [512,512,3,3]
    const float* mod_w    = (const float*)d_in[3];   // [512,512]
    const float* mod_b    = (const float*)d_in[4];   // [512]
    float*       out      = (float*)d_out;           // [16,512,64,64]

    // workspace carve-up (floats first for alignment)
    float* style = (float*)d_ws;                 // 16*512
    float* s2    = style + 16 * IN_CH;           // 512*512
    float* demod = s2 + OUT_CH * IN_CH;          // 16*512
    bf16*  Wr    = (bf16*)(demod + 16 * OUT_CH); // 9*512*512 bf16

    mc_style_kernel<<<(16 * IN_CH) / 256, 256, 0, stream>>>(w, mod_w, mod_b, style);
    mc_s2_kernel<<<(OUT_CH * IN_CH) / 256, 256, 0, stream>>>(weight, s2);
    mc_demod_kernel<<<(16 * OUT_CH) / 256, 256, 0, stream>>>(style, s2, demod);
    mc_repack_kernel<<<(KTAPS * OUT_CH * IN_CH) / 256, 256, 0, stream>>>(weight, Wr);

    dim3 grid(NPIX / 256, OUT_CH / 128, 16);
    mc_conv_kernel<<<grid, 256, 0, stream>>>(x, Wr, style, demod, out);
}
